// FusedMoEExperts_32023276159040
// MI455X (gfx1250) — matrix operand order repaired and checked by reading, not compiled, not run on hardware
//
#include <hip/hip_runtime.h>
#include <hip/hip_bf16.h>

typedef _Float16 half_t;
typedef __attribute__((ext_vector_type(16))) _Float16 v16h;
typedef __attribute__((ext_vector_type(8)))  _Float16 h8;
typedef __attribute__((ext_vector_type(4)))  _Float16 h4;
typedef __attribute__((ext_vector_type(2)))  _Float16 h2;
typedef __attribute__((ext_vector_type(8)))  float    v8f;

#define T_TOK 4096
#define HDIM  2048
#define IDIM  1024
#define NEXP  8
#define TOPK  2

// ---------------------------------------------------------------------------
// LDS -> WMMA fragment loaders, per CDNA5 ISA §7.12.2 (wave32).
// A (16x32 f16): lane L holds row M=L%16; K runs: e=0..7 -> K=g*8+e,
//                e=8..15 -> K=16+g*8+(e-8), g = L/16.  Two 16B ds_load_b128.
// B (32x16 f16): lane L holds col N=L%16; same K runs as A. Two 16B ds_load_b128.
// C/D (16x16 f32): VGPR r: lanes 0-15 -> M=r, lanes 16-31 -> M=r+8; N=L%16.
// ---------------------------------------------------------------------------
__device__ __forceinline__ v16h load_a_frag(const half_t* sA, int lane, int kbase) {
  const int m = lane & 15, g = lane >> 4;
  const half_t* p = sA + m * 64 + kbase + g * 8;
  h8 lo = *(const h8*)(p);
  h8 hi = *(const h8*)(p + 16);
  return __builtin_shufflevector(lo, hi, 0,1,2,3,4,5,6,7,8,9,10,11,12,13,14,15);
}

__device__ __forceinline__ v16h load_b_frag(const half_t* sW, int lane, int kbase, int nbase) {
  const int n = lane & 15, g = lane >> 4;
  const half_t* p = sW + (nbase + n) * 64 + kbase + g * 8;
  h8 lo = *(const h8*)(p);
  h8 hi = *(const h8*)(p + 16);
  return __builtin_shufflevector(lo, hi, 0,1,2,3,4,5,6,7,8,9,10,11,12,13,14,15);
}

// ---------------------------------------------------------------------------
// Packed int4 dequant via the f16 exponent trick:
//   0x6400 | n  ==  (half)(1024 + n)  exactly, for n in [0,15].
//   (n - 8) * s == (1024 + n) * s + (-1032 * s)   -> one v_pk_fma_f16.
// ---------------------------------------------------------------------------
__device__ __forceinline__ h2 dq2(int v, h2 s2, h2 c2) {
  const unsigned t = ((unsigned)(v & 15)) | (((unsigned)(v & 0xF0)) << 12) | 0x64006400u;
  const h2 hv = __builtin_bit_cast(h2, t);
  return hv * s2 + c2;   // contracts to v_pk_fma_f16
}

// ---------------------------------------------------------------------------
// Kernel 0: zero output (float4 stores) + expert counters.
// ---------------------------------------------------------------------------
__global__ void moe_init(float4* __restrict__ out4, int* __restrict__ counts, int n4) {
  const int stride = gridDim.x * blockDim.x;
  int i = blockIdx.x * blockDim.x + threadIdx.x;
  if (i < NEXP) counts[i] = 0;
  const float4 z = {0.0f, 0.0f, 0.0f, 0.0f};
  for (; i < n4; i += stride) out4[i] = z;
}

// ---------------------------------------------------------------------------
// Kernel 1: routing — build per-expert token lists (entries are t*TOPK+k).
// ---------------------------------------------------------------------------
__global__ void moe_route(const int* __restrict__ eidx, int* __restrict__ counts,
                          int* __restrict__ tlist) {
  const int tk = blockIdx.x * blockDim.x + threadIdx.x;
  if (tk >= T_TOK * TOPK) return;
  const int e = eidx[tk];
  const int pos = atomicAdd(counts + e, 1);
  tlist[e * T_TOK + pos] = tk;
}

// ---------------------------------------------------------------------------
// Kernel 2: h = silu(x @ W1g^T) * (x @ W1u^T) for routed tokens.
// Block: 32 tokens x 128 intermediate cols, 8 waves; each wave owns two
// 16x16 M-subtiles of gate AND up (8 WMMAs per 64-wide K chunk), so the
// fixed per-chunk weight-dequant VALU cost is amortized over 2x the tokens.
// ---------------------------------------------------------------------------
__global__ __launch_bounds__(256) void moe_gemm1_silu(
    const float* __restrict__ x, const int* __restrict__ w1,
    const float* __restrict__ w1s, const int* __restrict__ counts,
    const int* __restrict__ tlist, half_t* __restrict__ hbuf) {
  const int e = blockIdx.x >> 7;        // 128 token tiles (of 32) per expert
  const int tile = blockIdx.x & 127;
  const int cnt = counts[e];
  if (tile * 32 >= cnt) return;         // uniform early-exit
  const int i0 = blockIdx.y * 128;
  const int tid = threadIdx.x, lane = tid & 31, wave = tid >> 5;

  __shared__ __attribute__((aligned(16))) half_t sAx[32 * 64];
  __shared__ __attribute__((aligned(16))) half_t sWg[128 * 64];
  __shared__ __attribute__((aligned(16))) half_t sWu[128 * 64];
  __shared__ int sTok[32];

  if (tid < 32) {
    const int idx = tile * 32 + tid;
    sTok[tid] = (idx < cnt) ? tlist[e * T_TOK + idx] : -1;
  }
  __syncthreads();

  // x staging: thread -> (row = tid/8, 8 consecutive h)
  const int arow = tid >> 3, acol = (tid & 7) * 8;
  int tokA = sTok[arow]; if (tokA < 0) tokA = sTok[0];   // dummy row, masked later
  const float* xrow = x + (size_t)(tokA >> 1) * HDIM;

  // weight staging: thread -> (row = tid/2, 16 packed bytes = 32 h = 1 group)
  const int wrow = tid >> 1, wsel = tid & 1;
  const int*   pg  = w1  + ((size_t)e * (2 * IDIM) + (i0 + wrow)) * (HDIM / 2) + wsel * 16;
  const int*   pu  = w1  + ((size_t)e * (2 * IDIM) + (IDIM + i0 + wrow)) * (HDIM / 2) + wsel * 16;
  const float* psg = w1s + ((size_t)e * (2 * IDIM) + (i0 + wrow)) * (HDIM / 32) + wsel;
  const float* psu = w1s + ((size_t)e * (2 * IDIM) + (IDIM + i0 + wrow)) * (HDIM / 32) + wsel;

  v8f accg0 = {}, accg1 = {}, accu0 = {}, accu1 = {};

  for (int h0 = 0; h0 < HDIM; h0 += 64) {
    // ---- global loads for this chunk (before barrier) ----
    const float4 xf0 = *(const float4*)(xrow + h0 + acol);
    const float4 xf1 = *(const float4*)(xrow + h0 + acol + 4);
    int qg[16], qu[16];
    {
      const int4* g4 = (const int4*)(pg + (h0 >> 1));
      const int4* u4 = (const int4*)(pu + (h0 >> 1));
#pragma unroll
      for (int j = 0; j < 4; ++j) {
        int4 a = g4[j]; qg[4*j] = a.x; qg[4*j+1] = a.y; qg[4*j+2] = a.z; qg[4*j+3] = a.w;
        int4 b = u4[j]; qu[4*j] = b.x; qu[4*j+1] = b.y; qu[4*j+2] = b.z; qu[4*j+3] = b.w;
      }
    }
    const float sg = psg[h0 >> 5];
    const float su = psu[h0 >> 5];
    const h2 sg2 = { (half_t)sg, (half_t)sg };
    const h2 cg2 = { (half_t)(-1032.0f * sg), (half_t)(-1032.0f * sg) };
    const h2 su2 = { (half_t)su, (half_t)su };
    const h2 cu2 = { (half_t)(-1032.0f * su), (half_t)(-1032.0f * su) };
    if (h0 + 64 < HDIM) {   // prefetch next weight chunk (L2-resident stream)
      __builtin_prefetch(pg + ((h0 + 64) >> 1), 0, 3);
      __builtin_prefetch(pu + ((h0 + 64) >> 1), 0, 3);
    }

    __syncthreads();   // previous chunk's LDS reads done
    // ---- stage x tile as f16 (one 16B ds_store per thread) ----
    h8 hv = { (half_t)xf0.x, (half_t)xf0.y, (half_t)xf0.z, (half_t)xf0.w,
              (half_t)xf1.x, (half_t)xf1.y, (half_t)xf1.z, (half_t)xf1.w };
    *(h8*)(sAx + arow * 64 + acol) = hv;
    // ---- packed dequant int4 -> f16 into LDS (merges into ds_store_b128) ----
    const int kb = wsel * 32;
#pragma unroll
    for (int j = 0; j < 16; ++j) {
      *(h2*)(sWg + wrow * 64 + kb + 2 * j) = dq2(qg[j], sg2, cg2);
      *(h2*)(sWu + wrow * 64 + kb + 2 * j) = dq2(qu[j], su2, cu2);
    }
    __syncthreads();

    // ---- 2 k-steps x 2 M-subtiles x (gate, up) = 8 WMMAs ----
#pragma unroll
    for (int ks = 0; ks < 2; ++ks) {
      v16h a0 = load_a_frag(sAx,            lane, ks * 32);
      v16h a1 = load_a_frag(sAx + 16 * 64,  lane, ks * 32);
      v16h bg = load_b_frag(sWg, lane, ks * 32, wave * 16);
      v16h bu = load_b_frag(sWu, lane, ks * 32, wave * 16);
      accg0 = __builtin_amdgcn_wmma_f32_16x16x32_f16(false, a0, false, bg,
                                                     (short)0, accg0, false, false);
      accg1 = __builtin_amdgcn_wmma_f32_16x16x32_f16(false, a1, false, bg,
                                                     (short)0, accg1, false, false);
      accu0 = __builtin_amdgcn_wmma_f32_16x16x32_f16(false, a0, false, bu,
                                                     (short)0, accu0, false, false);
      accu1 = __builtin_amdgcn_wmma_f32_16x16x32_f16(false, a1, false, bu,
                                                     (short)0, accu1, false, false);
    }
  }

  // ---- fused SiLU(gate)*up epilogue, write h (f16) ----
  const int ncol = i0 + wave * 16 + (lane & 15);
  const int mhi = (lane >> 4) * 8;
#pragma unroll
  for (int sub = 0; sub < 2; ++sub) {
    const v8f& ag = sub ? accg1 : accg0;
    const v8f& au = sub ? accu1 : accu0;
#pragma unroll
    for (int r = 0; r < 8; ++r) {
      const int m = sub * 16 + mhi + r;
      const int tk = sTok[m];
      if (tk >= 0) {
        const float g = ag[r], u = au[r];
        const float s = g / (1.0f + __expf(-g));
        hbuf[(size_t)tk * IDIM + ncol] = (half_t)(s * u);
      }
    }
  }
}

// ---------------------------------------------------------------------------
// Kernel 3: y = h @ W2^T, scaled by softmax expert weight, atomically
// accumulated into out.  32-token tiles; the f16 h-tile gather goes straight
// to LDS via GLOBAL_LOAD_ASYNC_TO_LDS_B128 (ASYNCcnt) + s_wait_asynccnt.
// ---------------------------------------------------------------------------
__global__ __launch_bounds__(256) void moe_gemm2_scatter(
    const int* __restrict__ w2, const float* __restrict__ w2s,
    const float* __restrict__ ew, const int* __restrict__ counts,
    const int* __restrict__ tlist, const half_t* __restrict__ hbuf,
    float* __restrict__ out) {
  const int e = blockIdx.x >> 7;
  const int tile = blockIdx.x & 127;
  const int cnt = counts[e];
  if (tile * 32 >= cnt) return;
  const int hc0 = blockIdx.y * 128;
  const int tid = threadIdx.x, lane = tid & 31, wave = tid >> 5;

  __shared__ __attribute__((aligned(16))) half_t sAh[32 * 64];
  __shared__ __attribute__((aligned(16))) half_t sW[128 * 64];
  __shared__ int sTok[32];
  __shared__ float sEw[32];

  if (tid < 32) {
    const int idx = tile * 32 + tid;
    const int tk = (idx < cnt) ? tlist[e * T_TOK + idx] : -1;
    sTok[tid] = tk;
    sEw[tid] = (tk >= 0) ? ew[tk] : 0.0f;
  }
  __syncthreads();

  // h staging: thread -> (row = tid/8, 8 consecutive i) = 16B per lane
  const int arow = tid >> 3, acol = (tid & 7) * 8;
  int tokA = sTok[arow]; if (tokA < 0) tokA = sTok[0];
  const half_t* hrow = hbuf + (size_t)tokA * IDIM;
  // LDS aperture maps generic addr[31:0] -> LDS offset (ISA §10.2), so the
  // low 32 bits of the generic shared pointer are the dsaddr for async ops.
  const unsigned ldsA = (unsigned)(size_t)(sAh + arow * 64 + acol);

  const int wrow = tid >> 1, wsel = tid & 1;
  const int*   pw  = w2  + ((size_t)e * HDIM + (hc0 + wrow)) * (IDIM / 2) + wsel * 16;
  const float* psw = w2s + ((size_t)e * HDIM + (hc0 + wrow)) * (IDIM / 32) + wsel;

  v8f acc0 = {}, acc1 = {};

  for (int ic = 0; ic < IDIM; ic += 64) {
    int q[16];
    const int4* q4 = (const int4*)(pw + (ic >> 1));
#pragma unroll
    for (int j = 0; j < 4; ++j) {
      int4 a = q4[j]; q[4*j] = a.x; q[4*j+1] = a.y; q[4*j+2] = a.z; q[4*j+3] = a.w;
    }
    const float sc = psw[ic >> 5];
    const h2 s2 = { (half_t)sc, (half_t)sc };
    const h2 c2 = { (half_t)(-1032.0f * sc), (half_t)(-1032.0f * sc) };
    if (ic + 64 < IDIM) __builtin_prefetch(pw + ((ic + 64) >> 1), 0, 3);

    __syncthreads();   // previous chunk's LDS reads done
    // ---- async copy h tile (16B per lane) directly into LDS ----
    {
      const unsigned long long ga = (unsigned long long)(size_t)(hrow + ic + acol);
      asm volatile("global_load_async_to_lds_b128 %0, %1, off"
                   :: "v"(ldsA), "v"(ga) : "memory");
    }
    // ---- packed dequant int4 -> f16 into LDS ----
    const int kb = wsel * 32;
#pragma unroll
    for (int j = 0; j < 16; ++j) {
      *(h2*)(sW + wrow * 64 + kb + 2 * j) = dq2(q[j], s2, c2);
    }
    asm volatile("s_wait_asynccnt 0" ::: "memory");
    __syncthreads();

#pragma unroll
    for (int ks = 0; ks < 2; ++ks) {
      v16h a0 = load_a_frag(sAh,           lane, ks * 32);
      v16h a1 = load_a_frag(sAh + 16 * 64, lane, ks * 32);
      v16h b  = load_b_frag(sW, lane, ks * 32, wave * 16);
      acc0 = __builtin_amdgcn_wmma_f32_16x16x32_f16(false, a0, false, b,
                                                    (short)0, acc0, false, false);
      acc1 = __builtin_amdgcn_wmma_f32_16x16x32_f16(false, a1, false, b,
                                                    (short)0, acc1, false, false);
    }
  }

  const int ncol = hc0 + wave * 16 + (lane & 15);
  const int mhi = (lane >> 4) * 8;
#pragma unroll
  for (int sub = 0; sub < 2; ++sub) {
    const v8f& ac = sub ? acc1 : acc0;
#pragma unroll
    for (int r = 0; r < 8; ++r) {
      const int m = sub * 16 + mhi + r;
      const int tk = sTok[m];
      if (tk >= 0) {
        const int t = tk >> 1;
        atomicAdd(out + (size_t)t * HDIM + ncol, ac[r] * sEw[m]);
      }
    }
  }
}

// ---------------------------------------------------------------------------
// Workspace layout: [counts: 8 ints][pad to 256B][tlist: E*T ints = 128KB]
//                   [hbuf: T*K*I f16 = 16MB]
// ---------------------------------------------------------------------------
extern "C" void kernel_launch(void* const* d_in, const int* in_sizes, int n_in,
                              void* d_out, int out_size, void* d_ws, size_t ws_size,
                              hipStream_t stream) {
  const float* x    = (const float*)d_in[0];
  const int*   w1   = (const int*)d_in[1];
  const float* w1s  = (const float*)d_in[2];
  const int*   w2   = (const int*)d_in[3];
  const float* w2s  = (const float*)d_in[4];
  const float* ew   = (const float*)d_in[5];
  const int*   eidx = (const int*)d_in[6];
  float* out = (float*)d_out;

  char* ws = (char*)d_ws;
  int* counts = (int*)ws;
  int* tlist  = (int*)(ws + 256);
  half_t* hbuf = (half_t*)(ws + 256 + (size_t)NEXP * T_TOK * sizeof(int));

  moe_init<<<1024, 256, 0, stream>>>((float4*)out, counts, (T_TOK * HDIM) / 4);
  moe_route<<<(T_TOK * TOPK + 255) / 256, 256, 0, stream>>>(eidx, counts, tlist);

  dim3 g1(NEXP * (T_TOK / 32), IDIM / 128);
  moe_gemm1_silu<<<g1, 256, 0, stream>>>(x, w1, w1s, counts, tlist, hbuf);

  dim3 g2(NEXP * (T_TOK / 32), HDIM / 128);
  moe_gemm2_scatter<<<g2, 256, 0, stream>>>(w2, w2s, ew, counts, tlist, hbuf, out);

  (void)in_sizes; (void)n_in; (void)out_size; (void)ws_size;
}
